// Attention_10445360464069
// MI455X (gfx1250) — matrix operand order repaired and checked by reading, not compiled, not run on hardware
//
#include <hip/hip_runtime.h>
#include <hip/hip_bf16.h>

// ---------------------------------------------------------------------------
// Attention layer for MI455X (gfx1250): all matmuls on v_wmma_f32_16x16x32_bf16
// (fp32 accumulate), fp32 RoPE + softmax. Compute-bound problem (~240 GFLOP vs
// ~8us of HBM traffic at 23.3 TB/s), so everything routes through the WMMA
// pipes in bf16. All weights/activations fit in the 192MB L2; GEMM waves use a
// 32x64 tile (8 WMMAs / k-step) for ~21 FLOP/B from L2, plus explicit
// global_prefetch_b8 lookahead on the streamed rows.
// ---------------------------------------------------------------------------

typedef __bf16 bf16_t;
typedef __attribute__((ext_vector_type(16))) __bf16 v16bf;
typedef __attribute__((ext_vector_type(8)))  __bf16 v8bf;
typedef __attribute__((ext_vector_type(8)))  float  v8f;

constexpr int S_    = 2048;
constexpr int D_    = 4096;
constexpr int H_    = 32;
constexpr int KVH_  = 8;
constexpr int HD_   = 128;
constexpr int HALF_ = 64;
// N_REP = H_/KVH_ = 4

__device__ __forceinline__ v8f zero8() {
  v8f z;
#pragma unroll
  for (int i = 0; i < 8; ++i) z[i] = 0.0f;
  return z;
}

// ---------------------------------------------------------------------------
// fp32 -> bf16 elementwise conversion
// ---------------------------------------------------------------------------
__global__ void cvt_f32_bf16(const float* __restrict__ in, bf16_t* __restrict__ out, int n) {
  int i = blockIdx.x * blockDim.x + threadIdx.x;
  if (i < n) out[i] = (bf16_t)in[i];
}

// ---------------------------------------------------------------------------
// C[M,N] = A[M,K] @ W[N,K]^T   (A,W bf16 row-major; C fp32 or bf16)
// 128 threads = 4 waves; each wave computes a 32(M) x 64(N) tile:
//   2 A frags x 4 B frags -> 8 WMMAs per k-step of 32.
// A frag: lane row = lane%16, K contig  [g*16..g*16+15] (= contiguous A row).
// B frag: lane col = lane%16, K contig  [g*16..g*16+15] (= contiguous W row).
// ---------------------------------------------------------------------------
template <typename OutT>
__global__ __launch_bounds__(128)
void gemm_bf16_wmma(const bf16_t* __restrict__ A, const bf16_t* __restrict__ W,
                    OutT* __restrict__ C, int M, int N, int K) {
  const int lane = threadIdx.x & 31;
  const int wave = threadIdx.x >> 5;
  const int n    = lane & 15;
  const int g    = lane >> 4;
  const int m0   = blockIdx.x * 128 + wave * 32;
  const int n0   = blockIdx.y * 64;

  v8f acc[2][4];
#pragma unroll
  for (int u = 0; u < 2; ++u)
#pragma unroll
    for (int t = 0; t < 4; ++t) acc[u][t] = zero8();

  const bf16_t* arow[2];
  arow[0] = A + (size_t)(m0 + n) * K;
  arow[1] = A + (size_t)(m0 + 16 + n) * K;
  const bf16_t* wrow[4];
#pragma unroll
  for (int t = 0; t < 4; ++t)
    wrow[t] = W + (size_t)(n0 + t * 16 + n) * K + g * 16;

  constexpr int PF = 256;  // prefetch lookahead: 8 k-steps = 512 bytes
  for (int k0 = 0; k0 < K; k0 += 32) {
    if (k0 + PF < K) {
      __builtin_prefetch(arow[0] + k0 + PF, 0, 1);
      __builtin_prefetch(arow[1] + k0 + PF, 0, 1);
#pragma unroll
      for (int t = 0; t < 4; ++t)
        __builtin_prefetch(wrow[t] + k0 + PF, 0, 1);
    }

    v16bf a[2];
#pragma unroll
    for (int u = 0; u < 2; ++u) {
      v8bf lo = *(const v8bf*)(arow[u] + k0 + g * 16);
      v8bf hi = *(const v8bf*)(arow[u] + k0 + g * 16 + 8);
#pragma unroll
      for (int i = 0; i < 8; ++i) { a[u][i] = lo[i]; a[u][8 + i] = hi[i]; }
    }
#pragma unroll
    for (int t = 0; t < 4; ++t) {
      v16bf b = *(const v16bf*)(wrow[t] + k0);
      acc[0][t] = __builtin_amdgcn_wmma_f32_16x16x32_bf16(
          false, a[0], false, b, (short)0, acc[0][t], false, false);
      acc[1][t] = __builtin_amdgcn_wmma_f32_16x16x32_bf16(
          false, a[1], false, b, (short)0, acc[1][t], false, false);
    }
  }

#pragma unroll
  for (int u = 0; u < 2; ++u)
#pragma unroll
    for (int t = 0; t < 4; ++t)
#pragma unroll
      for (int r = 0; r < 8; ++r) {
        size_t row = (size_t)(m0 + u * 16 + r + 8 * g);
        C[row * N + n0 + t * 16 + n] = (OutT)acc[u][t][r];
      }
}

// ---------------------------------------------------------------------------
// RoPE on Q (with 1/sqrt(HD) folded in) and K; fp32 in, bf16 out.
// One thread per (s, head, pair).
// ---------------------------------------------------------------------------
__global__ void rope_qk(const float* __restrict__ qf, const float* __restrict__ kf,
                        const float* __restrict__ cosp, const float* __restrict__ sinp,
                        bf16_t* __restrict__ qb, bf16_t* __restrict__ kb) {
  int idx  = blockIdx.x * blockDim.x + threadIdx.x;   // S*(H+KVH)*HALF threads
  int i    = idx % HALF_;
  int head = (idx / HALF_) % (H_ + KVH_);
  int s    = idx / (HALF_ * (H_ + KVH_));
  if (s >= S_) return;
  float c  = cosp[s * HALF_ + i];
  float sn = sinp[s * HALF_ + i];
  if (head < H_) {
    const float scale = 0.08838834764831845f;  // 1/sqrt(128)
    const float* p = qf + ((size_t)s * H_ + head) * HD_ + 2 * i;
    bf16_t* o = qb + ((size_t)s * H_ + head) * HD_ + 2 * i;
    float te = p[0], to = p[1];
    o[0] = (bf16_t)((te * c - to * sn) * scale);
    o[1] = (bf16_t)((te * sn + to * c) * scale);
  } else {
    int kvh = head - H_;
    const float* p = kf + ((size_t)s * KVH_ + kvh) * HD_ + 2 * i;
    bf16_t* o = kb + ((size_t)s * KVH_ + kvh) * HD_ + 2 * i;
    float te = p[0], to = p[1];
    o[0] = (bf16_t)(te * c - to * sn);
    o[1] = (bf16_t)(te * sn + to * c);
  }
}

// ---------------------------------------------------------------------------
// V transpose: vt[kvh][hd][s] = (bf16) v[s][kvh][hd]
// ---------------------------------------------------------------------------
__global__ void vtrans(const float* __restrict__ vf, bf16_t* __restrict__ vt) {
  int idx = blockIdx.x * blockDim.x + threadIdx.x;   // S*KVH*HD threads
  int hd  = idx % HD_;
  int kvh = (idx / HD_) % KVH_;
  int s   = idx / (HD_ * KVH_);
  if (s >= S_) return;
  vt[((size_t)kvh * HD_ + hd) * S_ + s] = (bf16_t)vf[idx];
}

// ---------------------------------------------------------------------------
// Causal flash attention, bf16 WMMA matmuls, fp32 online softmax.
// Grid: (S/64, H). Block: 128 threads = 4 waves, each wave = one 16-row Q tile
// of the same head. Uniform per-block trip count -> __syncthreads is legal.
// P (scores) is bounced through LDS to convert C/D layout -> A layout.
// ---------------------------------------------------------------------------
__global__ __launch_bounds__(128)
void flash_attn(const bf16_t* __restrict__ qb, const bf16_t* __restrict__ kb,
                const bf16_t* __restrict__ vt, bf16_t* __restrict__ ab) {
  __shared__ __bf16 plds[4][16 * 32];

  const int lane = threadIdx.x & 31;
  const int wave = threadIdx.x >> 5;
  const int n    = lane & 15;
  const int g    = lane >> 4;
  const int h    = blockIdx.y;
  const int kvh  = h >> 2;              // h / N_REP
  const int q0b  = blockIdx.x * 64;
  const int q0   = q0b + wave * 16;

  // Q fragments: 4 chunks of K=32 over HD=128
  v16bf qa[4];
  {
    const bf16_t* qrow = qb + ((size_t)(q0 + n) * H_ + h) * HD_;
#pragma unroll
    for (int c = 0; c < 4; ++c) {
      v8bf lo = *(const v8bf*)(qrow + c * 32 + g * 16);
      v8bf hi = *(const v8bf*)(qrow + c * 32 + g * 16 + 8);
#pragma unroll
      for (int i = 0; i < 8; ++i) { qa[c][i] = lo[i]; qa[c][8 + i] = hi[i]; }
    }
  }

  v8f oacc[8];
#pragma unroll
  for (int t = 0; t < 8; ++t) oacc[t] = zero8();
  float mrow[8], lrow[8];
#pragma unroll
  for (int r = 0; r < 8; ++r) { mrow[r] = -1e30f; lrow[r] = 0.0f; }

  const int ntiles = q0b / 32 + 2;      // keys 0 .. q0b+63 in tiles of 32

  for (int j = 0; j < ntiles; ++j) {
    const int t0 = j * 32;

    // ---- scores: S = Q @ K^T for 16 rows x 32 keys (two 16x16 blocks) ----
    v8f sc[2];
#pragma unroll
    for (int half = 0; half < 2; ++half) {
      v8f s = zero8();
      const bf16_t* krow =
          kb + ((size_t)(t0 + half * 16 + n) * KVH_ + kvh) * HD_ + g * 16;
#pragma unroll
      for (int c = 0; c < 4; ++c) {
        v16bf b = *(const v16bf*)(krow + c * 32);
        s = __builtin_amdgcn_wmma_f32_16x16x32_bf16(
            false, qa[c], false, b, (short)0, s, false, false);
      }
      sc[half] = s;
    }

    // ---- causal mask + online softmax (per result row r+8g) ----
#pragma unroll
    for (int r = 0; r < 8; ++r) {
      const int qrow_i = q0 + r + 8 * g;
      float s0 = sc[0][r];
      float s1 = sc[1][r];
      if (t0 + n > qrow_i)      s0 = -1e30f;
      if (t0 + 16 + n > qrow_i) s1 = -1e30f;

      float mx = fmaxf(s0, s1);
#pragma unroll
      for (int off = 8; off >= 1; off >>= 1)
        mx = fmaxf(mx, __shfl_xor(mx, off, 32));
      const float mnew  = fmaxf(mrow[r], mx);
      const float scale = __expf(mrow[r] - mnew);
      const float p0 = __expf(s0 - mnew);
      const float p1 = __expf(s1 - mnew);
      float ps = p0 + p1;
#pragma unroll
      for (int off = 8; off >= 1; off >>= 1)
        ps += __shfl_xor(ps, off, 32);
      lrow[r] = lrow[r] * scale + ps;
      mrow[r] = mnew;
#pragma unroll
      for (int t = 0; t < 8; ++t) oacc[t][r] *= scale;

      plds[wave][(r + 8 * g) * 32 + n]      = (bf16_t)p0;
      plds[wave][(r + 8 * g) * 32 + 16 + n] = (bf16_t)p1;
    }
    __syncthreads();

    // ---- reload P in A-matrix layout ----
    v16bf pa;
    {
      const __bf16* pr = &plds[wave][n * 32];
      v8bf lo = *(const v8bf*)(pr + g * 16);
      v8bf hi = *(const v8bf*)(pr + g * 16 + 8);
#pragma unroll
      for (int i = 0; i < 8; ++i) { pa[i] = lo[i]; pa[8 + i] = hi[i]; }
    }
    __syncthreads();

    // ---- O += P @ V  (V^T rows are contiguous in time) ----
#pragma unroll
    for (int t = 0; t < 8; ++t) {
      const bf16_t* vrow =
          vt + ((size_t)kvh * HD_ + t * 16 + n) * S_ + t0 + g * 16;
      v16bf b = *(const v16bf*)vrow;
      oacc[t] = __builtin_amdgcn_wmma_f32_16x16x32_bf16(
          false, pa, false, b, (short)0, oacc[t], false, false);
    }
  }

  // ---- finalize: O / l, store bf16 into (S, H*HD) ----
#pragma unroll
  for (int r = 0; r < 8; ++r) {
    const float inv = 1.0f / lrow[r];
    bf16_t* outp =
        ab + (size_t)(q0 + r + 8 * g) * (H_ * HD_) + (size_t)h * HD_;
#pragma unroll
    for (int t = 0; t < 8; ++t)
      outp[t * 16 + n] = (bf16_t)(oacc[t][r] * inv);
  }
}

// ---------------------------------------------------------------------------
// Host-side orchestration
// ---------------------------------------------------------------------------
extern "C" void kernel_launch(void* const* d_in, const int* in_sizes, int n_in,
                              void* d_out, int out_size, void* d_ws, size_t ws_size,
                              hipStream_t stream) {
  const float* x    = (const float*)d_in[0];
  const float* wq   = (const float*)d_in[1];
  const float* wk   = (const float*)d_in[2];
  const float* wv   = (const float*)d_in[3];
  const float* wo   = (const float*)d_in[4];
  const float* cosp = (const float*)d_in[5];
  const float* sinp = (const float*)d_in[6];
  (void)in_sizes; (void)n_in; (void)ws_size; (void)out_size; // mask/start_pos unused

  char* ws = (char*)d_ws;
  size_t off = 0;
  auto alloc = [&](size_t bytes) -> void* {
    off = (off + 255) & ~(size_t)255;
    void* p = ws + off;
    off += bytes;
    return p;
  };

  const size_t SD  = (size_t)S_ * D_;        // 2048*4096
  const size_t QN  = (size_t)H_ * HD_;       // 4096
  const size_t KVN = (size_t)KVH_ * HD_;     // 1024

  bf16_t* xb  = (bf16_t*)alloc(SD * 2);
  bf16_t* wqb = (bf16_t*)alloc(QN * D_ * 2);
  bf16_t* wkb = (bf16_t*)alloc(KVN * D_ * 2);
  bf16_t* wvb = (bf16_t*)alloc(KVN * D_ * 2);
  bf16_t* wob = (bf16_t*)alloc((size_t)D_ * QN * 2);
  float*  qf  = (float*)alloc((size_t)S_ * QN * 4);
  float*  kf  = (float*)alloc((size_t)S_ * KVN * 4);
  float*  vf  = (float*)alloc((size_t)S_ * KVN * 4);
  bf16_t* qb  = (bf16_t*)alloc((size_t)S_ * QN * 2);
  bf16_t* kb  = (bf16_t*)alloc((size_t)S_ * KVN * 2);
  bf16_t* vtb = (bf16_t*)alloc((size_t)S_ * KVN * 2);
  bf16_t* ab  = (bf16_t*)alloc((size_t)S_ * QN * 2);

  const int T = 256;
  auto blks = [&](size_t n) { return (unsigned)((n + T - 1) / T); };

  // 1) fp32 -> bf16 conversions
  cvt_f32_bf16<<<blks(SD), T, 0, stream>>>(x, xb, (int)SD);
  cvt_f32_bf16<<<blks(QN * D_), T, 0, stream>>>(wq, wqb, (int)(QN * D_));
  cvt_f32_bf16<<<blks(KVN * D_), T, 0, stream>>>(wk, wkb, (int)(KVN * D_));
  cvt_f32_bf16<<<blks(KVN * D_), T, 0, stream>>>(wv, wvb, (int)(KVN * D_));
  cvt_f32_bf16<<<blks((size_t)D_ * QN), T, 0, stream>>>(wo, wob, (int)((size_t)D_ * QN));

  // 2) QKV projections (C = x @ W^T), fp32 out
  gemm_bf16_wmma<float><<<dim3(S_ / 128, QN / 64), 128, 0, stream>>>(
      xb, wqb, qf, S_, (int)QN, D_);
  gemm_bf16_wmma<float><<<dim3(S_ / 128, KVN / 64), 128, 0, stream>>>(
      xb, wkb, kf, S_, (int)KVN, D_);
  gemm_bf16_wmma<float><<<dim3(S_ / 128, KVN / 64), 128, 0, stream>>>(
      xb, wvb, vf, S_, (int)KVN, D_);

  // 3) RoPE (Q scaled by 1/sqrt(HD)) -> bf16
  rope_qk<<<blks((size_t)S_ * (H_ + KVH_) * HALF_), T, 0, stream>>>(
      qf, kf, cosp, sinp, qb, kb);

  // 4) V transpose -> bf16 [kvh][hd][s]
  vtrans<<<blks((size_t)S_ * KVN), T, 0, stream>>>(vf, vtb);

  // 5) causal flash attention -> bf16 (S, H*HD)
  flash_attn<<<dim3(S_ / 64, H_), 128, 0, stream>>>(qb, kb, vtb, ab);

  // 6) output projection -> fp32 d_out
  gemm_bf16_wmma<float><<<dim3(S_ / 128, D_ / 64), 128, 0, stream>>>(
      ab, wob, (float*)d_out, S_, D_, (int)QN);
}
